// SparseMOELayer_57621281243506
// MI455X (gfx1250) — matrix operand order repaired and checked by reading, not compiled, not run on hardware
//
#include <hip/hip_runtime.h>
#include <hip/hip_bf16.h>

#define D_DIM 768
#define H_DIM 3072
#define E_NUM 8
#define N_TOK 32768
#define CAP   10240
#define LN_EPS 1e-5f

typedef __attribute__((ext_vector_type(16))) __bf16 v16bf;
typedef __attribute__((ext_vector_type(8)))  float  v8f;

union Frag16 { uint4 q[2]; v16bf v; };

__device__ inline unsigned short f2bf(float f) {
  unsigned u = __float_as_uint(f);
  unsigned r = ((u >> 16) & 1u) + 0x7FFFu;   // round-to-nearest-even
  return (unsigned short)((u + r) >> 16);
}
__device__ inline unsigned pk2bf(float lo, float hi) {
#if __has_builtin(__builtin_amdgcn_cvt_pk_bf16_f32)
  typedef __attribute__((ext_vector_type(2))) __bf16 v2bf;
  union { v2bf v; unsigned u; } c;
  c.v = __builtin_amdgcn_cvt_pk_bf16_f32(lo, hi);
  return c.u;
#else
  return (unsigned)f2bf(lo) | ((unsigned)f2bf(hi) << 16);
#endif
}

// ---------------------------------------------------------------- init
__global__ __launch_bounds__(256) void moe_init(float* __restrict__ out,
                                                int* __restrict__ cnt,
                                                int* __restrict__ tok,
                                                float* __restrict__ wt) {
  int i = blockIdx.x * 256 + threadIdx.x;
  if (i < N_TOK * D_DIM) out[i] = 0.f;
  if (i < E_NUM * CAP) { tok[i] = 0; wt[i] = 0.f; }
  if (i < E_NUM) cnt[i] = 0;
}

// ---------------------------------------------------------------- router
__global__ __launch_bounds__(256) void router_kernel(const float* __restrict__ x,
                                                     const float* __restrict__ wg,
                                                     int* __restrict__ cnt,
                                                     int* __restrict__ tok,
                                                     float* __restrict__ wt) {
  extern __shared__ float sW[];          // [D][E] = 24 KB
  const int tid = threadIdx.x;
  for (int i = tid; i < D_DIM * E_NUM; i += 256) sW[i] = wg[i];
  __syncthreads();

  const int t = blockIdx.x * 256 + tid;
  const float4* xp = (const float4*)(x + (size_t)t * D_DIM);
  float acc[E_NUM];
#pragma unroll
  for (int e = 0; e < E_NUM; ++e) acc[e] = 0.f;

  for (int d4 = 0; d4 < D_DIM / 4; ++d4) {
    float4 xv = xp[d4];
    const float* wr = &sW[d4 * 4 * E_NUM];     // broadcast LDS reads
#pragma unroll
    for (int e = 0; e < E_NUM; ++e)
      acc[e] += xv.x * wr[e] + xv.y * wr[8 + e] + xv.z * wr[16 + e] + xv.w * wr[24 + e];
  }

  int i1 = 0; float m1 = acc[0];
#pragma unroll
  for (int e = 1; e < E_NUM; ++e) if (acc[e] > m1) { m1 = acc[e]; i1 = e; }
  int i2 = -1; float m2 = -3.4e38f;
#pragma unroll
  for (int e = 0; e < E_NUM; ++e) if (e != i1 && acc[e] > m2) { m2 = acc[e]; i2 = e; }

  float e2 = __expf(m2 - m1);
  float inv = 1.f / (1.f + e2);
  float wA = inv, wB = e2 * inv;     // softmax over the two kept logits

  int p1 = atomicAdd(&cnt[i1], 1);
  if (p1 < CAP) { tok[i1 * CAP + p1] = t; wt[i1 * CAP + p1] = wA; }
  int p2 = atomicAdd(&cnt[i2], 1);
  if (p2 < CAP) { tok[i2 * CAP + p2] = t; wt[i2 * CAP + p2] = wB; }
}

// ---------------------------------------------------------------- GEMM1: H = gelu(X_e @ W1 + b1)
__global__ __launch_bounds__(256) void expert_ffn1(const float* __restrict__ x,
                                                   const float* __restrict__ w1,
                                                   const float* __restrict__ b1,
                                                   const int* __restrict__ tok,
                                                   unsigned short* __restrict__ hbuf,
                                                   int e) {
  extern __shared__ char smem[];                 // 20 KB
  unsigned*       sA  = (unsigned*)smem;                 // A: 128 rows * 20 dwords (80B stride)
  unsigned*       sBu = (unsigned*)(smem + 10240);       // B: 128 cols * 20 dwords
  unsigned short* sBh = (unsigned short*)(smem + 10240);

  const int tid = threadIdx.x;
  const int m_base = blockIdx.x * 128;
  const int n_base = blockIdx.y * 128;
  const int wave = tid >> 5, lane = tid & 31, lo = lane & 15, hi = lane >> 4;

  // A loader: 2 threads per row, 16 K-elems each
  const int arow = tid >> 1, ahalf = tid & 1;
  const int atok = tok[e * CAP + m_base + arow];
  const float* asrc = x + (size_t)atok * D_DIM + ahalf * 16;

  // B loader: thread -> (k row, 16 contiguous cols)
  const int bkk = tid >> 3;
  const int bcs = (tid & 7) * 16;
  const float* bsrc = w1 + (size_t)e * D_DIM * H_DIM + n_base + bcs;

  v8f zero = {};
  v8f acc[8];
#pragma unroll
  for (int i = 0; i < 8; ++i) acc[i] = zero;

  for (int k0 = 0; k0 < D_DIM; k0 += 32) {
    // ---- stage A (gathered x rows, f32 -> bf16)
    const float4* ap = (const float4*)(asrc + k0);
    float4 f0 = ap[0], f1 = ap[1], f2 = ap[2], f3 = ap[3];
    uint4 q0, q1;
    q0.x = pk2bf(f0.x, f0.y); q0.y = pk2bf(f0.z, f0.w);
    q0.z = pk2bf(f1.x, f1.y); q0.w = pk2bf(f1.z, f1.w);
    q1.x = pk2bf(f2.x, f2.y); q1.y = pk2bf(f2.z, f2.w);
    q1.z = pk2bf(f3.x, f3.y); q1.w = pk2bf(f3.z, f3.w);
    *(uint4*)&sA[arow * 20 + ahalf * 8]     = q0;
    *(uint4*)&sA[arow * 20 + ahalf * 8 + 4] = q1;

    // ---- stage B (w1 tile, transpose into [col][k], f32 -> bf16)
    const float* brow = bsrc + (size_t)(k0 + bkk) * H_DIM;
    float4 g0 = ((const float4*)brow)[0];
    float4 g1 = ((const float4*)brow)[1];
    float4 g2 = ((const float4*)brow)[2];
    float4 g3 = ((const float4*)brow)[3];
    float bv[16] = {g0.x,g0.y,g0.z,g0.w, g1.x,g1.y,g1.z,g1.w,
                    g2.x,g2.y,g2.z,g2.w, g3.x,g3.y,g3.z,g3.w};
#pragma unroll
    for (int i = 0; i < 16; ++i) sBh[(bcs + i) * 40 + bkk] = f2bf(bv[i]);

    if (k0 + 32 < D_DIM) {                      // gfx1250 global_prefetch_b8
      __builtin_prefetch(asrc + k0 + 32, 0, 1);
      __builtin_prefetch(bsrc + (size_t)(k0 + 32 + bkk) * H_DIM, 0, 1);
    }
    __syncthreads();

    // ---- compute: preload A + all 8 B frags, then 8 back-to-back WMMAs
    Frag16 a, bfrag[8];
    int aoff = (wave * 16 + lo) * 20 + hi * 4;
    a.q[0] = *(const uint4*)&sA[aoff];
    a.q[1] = *(const uint4*)&sA[aoff + 8];
#pragma unroll
    for (int nt = 0; nt < 8; ++nt) {
      int boff = (nt * 16 + lo) * 20 + hi * 4;
      bfrag[nt].q[0] = *(const uint4*)&sBu[boff];
      bfrag[nt].q[1] = *(const uint4*)&sBu[boff + 8];
    }
#pragma unroll
    for (int nt = 0; nt < 8; ++nt)
      acc[nt] = __builtin_amdgcn_wmma_f32_16x16x32_bf16(
          false, a.v, false, bfrag[nt].v, (short)0, acc[nt], false, false);
    __syncthreads();
  }

  // ---- epilogue: bias + exact GELU -> bf16 h_buf
  const float* b1p = b1 + e * H_DIM + n_base;
#pragma unroll
  for (int r = 0; r < 8; ++r) {
    int slot = m_base + wave * 16 + r + hi * 8;
    size_t rowoff = (size_t)slot * H_DIM + n_base;
#pragma unroll
    for (int nt = 0; nt < 8; ++nt) {
      int col = nt * 16 + lo;
      float v = acc[nt][r] + b1p[col];
      float g = 0.5f * v * (1.f + erff(v * 0.70710678118f));
      hbuf[rowoff + col] = f2bf(g);
    }
  }
}

// ---------------------------------------------------------------- GEMM2: out += (H @ W2 + b2) * w
__global__ __launch_bounds__(256) void expert_ffn2(const unsigned short* __restrict__ hbuf,
                                                   const float* __restrict__ w2,
                                                   const float* __restrict__ b2,
                                                   const int* __restrict__ tok,
                                                   const float* __restrict__ wt,
                                                   float* __restrict__ out,
                                                   int e) {
  extern __shared__ char smem[];
  unsigned*       sA  = (unsigned*)smem;
  unsigned*       sBu = (unsigned*)(smem + 10240);
  unsigned short* sBh = (unsigned short*)(smem + 10240);

  const int tid = threadIdx.x;
  const int m_base = blockIdx.x * 128;
  const int n_base = blockIdx.y * 128;
  const int wave = tid >> 5, lane = tid & 31, lo = lane & 15, hi = lane >> 4;

  const int arow = tid >> 1, ahalf = tid & 1;
  const unsigned short* asrc = hbuf + (size_t)(m_base + arow) * H_DIM + ahalf * 16;
  // LDS byte address for this thread's A destination (passed as the async load's vdst operand)
  const unsigned lds_a_off = (unsigned)(uintptr_t)(&sA[arow * 20 + ahalf * 8]);

  const int bkk = tid >> 3;
  const int bcs = (tid & 7) * 16;
  const float* bsrc = w2 + (size_t)e * H_DIM * D_DIM + n_base + bcs;

  v8f zero = {};
  v8f acc[8];
#pragma unroll
  for (int i = 0; i < 8; ++i) acc[i] = zero;

  for (int k0 = 0; k0 < H_DIM; k0 += 32) {
    // ---- stage A (already bf16): gfx1250 async global->LDS DMA, no VGPR round-trip
    {
      const unsigned short* ga = asrc + k0;
      asm volatile("global_load_async_to_lds_b128 %0, %1, off"
                   :: "v"(lds_a_off), "v"(ga) : "memory");
      asm volatile("global_load_async_to_lds_b128 %0, %1, off offset:16"
                   :: "v"(lds_a_off), "v"(ga) : "memory");
    }

    // ---- stage B (w2 tile transpose, f32 -> bf16)
    const float* brow = bsrc + (size_t)(k0 + bkk) * D_DIM;
    float4 g0 = ((const float4*)brow)[0];
    float4 g1 = ((const float4*)brow)[1];
    float4 g2 = ((const float4*)brow)[2];
    float4 g3 = ((const float4*)brow)[3];
    float bv[16] = {g0.x,g0.y,g0.z,g0.w, g1.x,g1.y,g1.z,g1.w,
                    g2.x,g2.y,g2.z,g2.w, g3.x,g3.y,g3.z,g3.w};
#pragma unroll
    for (int i = 0; i < 16; ++i) sBh[(bcs + i) * 40 + bkk] = f2bf(bv[i]);

    if (k0 + 32 < H_DIM) {
      __builtin_prefetch(asrc + k0 + 32, 0, 1);
      __builtin_prefetch(bsrc + (size_t)(k0 + 32 + bkk) * D_DIM, 0, 1);
    }
    asm volatile("s_wait_asynccnt 0x0" ::: "memory");   // A tile landed in LDS
    __syncthreads();

    Frag16 a, bfrag[8];
    int aoff = (wave * 16 + lo) * 20 + hi * 4;
    a.q[0] = *(const uint4*)&sA[aoff];
    a.q[1] = *(const uint4*)&sA[aoff + 8];
#pragma unroll
    for (int nt = 0; nt < 8; ++nt) {
      int boff = (nt * 16 + lo) * 20 + hi * 4;
      bfrag[nt].q[0] = *(const uint4*)&sBu[boff];
      bfrag[nt].q[1] = *(const uint4*)&sBu[boff + 8];
    }
#pragma unroll
    for (int nt = 0; nt < 8; ++nt)
      acc[nt] = __builtin_amdgcn_wmma_f32_16x16x32_bf16(
          false, a.v, false, bfrag[nt].v, (short)0, acc[nt], false, false);
    __syncthreads();
  }

  // ---- epilogue: weighted scatter-add (filler rows wt==0 skipped -> no races)
  const float* b2p = b2 + e * D_DIM + n_base;
#pragma unroll
  for (int r = 0; r < 8; ++r) {
    int slot = m_base + wave * 16 + r + hi * 8;
    int   tk = tok[e * CAP + slot];
    float w  = wt[e * CAP + slot];
    if (w != 0.f) {
      float* orow = out + (size_t)tk * D_DIM + n_base;
#pragma unroll
      for (int nt = 0; nt < 8; ++nt) {
        int col = nt * 16 + lo;
        float y = (acc[nt][r] + b2p[col]) * w;
        unsafeAtomicAdd(&orow[col], y);          // hw global_atomic_add_f32
      }
    }
  }
}

// ---------------------------------------------------------------- LayerNorm (in place)
__global__ __launch_bounds__(256) void ln_kernel(float* __restrict__ out,
                                                 const float* __restrict__ g,
                                                 const float* __restrict__ b) {
  __shared__ float s1[256], s2[256];
  const int tid = threadIdx.x;
  float* p = out + (size_t)blockIdx.x * D_DIM;
  float v0 = p[tid], v1 = p[tid + 256], v2 = p[tid + 512];
  s1[tid] = v0 + v1 + v2;
  s2[tid] = v0 * v0 + v1 * v1 + v2 * v2;
  __syncthreads();
  for (int o = 128; o > 0; o >>= 1) {
    if (tid < o) { s1[tid] += s1[tid + o]; s2[tid] += s2[tid + o]; }
    __syncthreads();
  }
  float mu  = s1[0] * (1.f / D_DIM);
  float var = s2[0] * (1.f / D_DIM) - mu * mu;
  float rs  = rsqrtf(var + LN_EPS);
  p[tid]       = (v0 - mu) * rs * g[tid]       + b[tid];
  p[tid + 256] = (v1 - mu) * rs * g[tid + 256] + b[tid + 256];
  p[tid + 512] = (v2 - mu) * rs * g[tid + 512] + b[tid + 512];
}

// ---------------------------------------------------------------- launch
extern "C" void kernel_launch(void* const* d_in, const int* in_sizes, int n_in,
                              void* d_out, int out_size, void* d_ws, size_t ws_size,
                              hipStream_t stream) {
  const float* x     = (const float*)d_in[0];
  const float* wgate = (const float*)d_in[1];
  const float* w1    = (const float*)d_in[2];
  const float* b1    = (const float*)d_in[3];
  const float* w2    = (const float*)d_in[4];
  const float* b2    = (const float*)d_in[5];
  const float* lnw   = (const float*)d_in[6];
  const float* lnb   = (const float*)d_in[7];
  float* out = (float*)d_out;

  char* ws = (char*)d_ws;
  int*            cnt  = (int*)ws;                                  // 32 B
  int*            tok  = (int*)(ws + 256);                          // E*CAP ints
  float*          wt   = (float*)(ws + 256 + E_NUM * CAP * 4);      // E*CAP floats
  unsigned short* hbuf = (unsigned short*)(ws + 256 + 2 * (size_t)E_NUM * CAP * 4);
  // hbuf: CAP * H_DIM bf16 = 60 MB (reused sequentially across experts)

  dim3 blk(256);
  moe_init<<<(N_TOK * D_DIM) / 256, blk, 0, stream>>>(out, cnt, tok, wt);
  router_kernel<<<N_TOK / 256, blk, D_DIM * E_NUM * 4, stream>>>(x, wgate, cnt, tok, wt);
  for (int e = 0; e < E_NUM; ++e) {
    expert_ffn1<<<dim3(CAP / 128, H_DIM / 128), blk, 20480, stream>>>(x, w1, b1, tok, hbuf, e);
    expert_ffn2<<<dim3(CAP / 128, D_DIM / 128), blk, 20480, stream>>>(hbuf, w2, b2, tok, wt, out, e);
  }
  ln_kernel<<<N_TOK, blk, 0, stream>>>(out, lnw, lnb);
}
